// MultiHeadAttention_90829968376179
// MI455X (gfx1250) — matrix operand order repaired and checked by reading, not compiled, not run on hardware
//
#include <hip/hip_runtime.h>
#include <hip/hip_bf16.h>

typedef __bf16 bf16_t;
typedef __attribute__((ext_vector_type(16))) __bf16 v16bf;
typedef __attribute__((ext_vector_type(8)))  __bf16 bf16x8;
typedef __attribute__((ext_vector_type(4)))  __bf16 bf16x4;
typedef __attribute__((ext_vector_type(8)))  float   v8f;
typedef __attribute__((ext_vector_type(4)))  unsigned int u32x4;
typedef __attribute__((ext_vector_type(8)))  int     i32x8;
typedef __attribute__((ext_vector_type(4)))  int     i32x4;

union Frag { v16bf v; bf16x8 h[2]; };

__device__ __forceinline__ v8f vzero8() {
    v8f z = {0.f,0.f,0.f,0.f,0.f,0.f,0.f,0.f};
    return z;
}

__device__ __forceinline__ v8f wmma_bf16(v16bf a, v16bf b, v8f c) {
    // D(f32 16x16) = A(bf16 16x32) * B(bf16 32x16) + C
    return __builtin_amdgcn_wmma_f32_16x16x32_bf16(
        /*neg_a=*/false, a, /*neg_b=*/false, b,
        /*c_mod=*/(short)0, c, /*reuse_a=*/false, /*reuse_b=*/false);
}

// LDS (addrspace 3) byte offset of a generic pointer to __shared__ data.
__device__ __forceinline__ unsigned int lds_byte_off(const void* p) {
    return (unsigned int)(unsigned long long)
        (__attribute__((address_space(3))) const char*)p;
}

// Problem constants (match reference)
#define BB   2
#define SS   2048
#define DD   1024
#define HH   16
#define HD   64
#define MM   (BB*SS)        // 4096 rows
// GEMM tiling
#define TM   128
#define TN   128
#define TK   32
#define PITCH 40            // bf16 units per LDS row (80B, 16B aligned, skews banks)
#define KPITCH 72           // K-tile pitch: 64 + 4 DWORDs of TDM padding

// ---------------------------------------------------------------------------
// Kernel 1: fused Q/K/V projections.  C[m,n] = sum_k X[m,k] * W[n,k]  (X @ W^T)
// fp32 inputs -> bf16 LDS tiles -> v_wmma_f32_16x16x32_bf16 -> bf16 workspace.
// grid: 3 gemms * (32 mtiles * 8 ntiles) = 768 blocks, 256 threads (8 waves).
// Each wave computes a 16(M) x 128(N) strip (8 accumulators).
// ---------------------------------------------------------------------------
__global__ __launch_bounds__(256) void qkv_proj_kernel(
    const float* __restrict__ q_in, const float* __restrict__ k_in,
    const float* __restrict__ v_in,
    const float* __restrict__ Wq, const float* __restrict__ Wk,
    const float* __restrict__ Wv,
    bf16_t* __restrict__ Qws, bf16_t* __restrict__ Kws, bf16_t* __restrict__ Vws)
{
    __shared__ alignas(16) bf16_t lA[TM * PITCH];
    __shared__ alignas(16) bf16_t lB[TN * PITCH];

    const int N = DD, K = DD;
    const int blocksPer = (MM/TM) * (N/TN);       // 32*8 = 256
    const int which = blockIdx.x / blocksPer;     // 0:Q 1:K 2:V
    const int t     = blockIdx.x % blocksPer;
    const int mt    = t / (N/TN);
    const int nt    = t % (N/TN);

    const float* X = (which == 0) ? q_in : (which == 1) ? k_in : v_in;
    const float* W = (which == 0) ? Wq   : (which == 1) ? Wk   : Wv;
    bf16_t*      O = (which == 0) ? Qws  : (which == 1) ? Kws  : Vws;

    const int tid  = threadIdx.x;
    const int lane = tid & 31;
    const int w    = tid >> 5;      // wave 0..7
    const int g    = lane >> 4;     // 16-lane group
    const int ln   = lane & 15;

    v8f acc[8];
#pragma unroll
    for (int s = 0; s < 8; ++s) acc[s] = vzero8();

    const int ldr = tid >> 3;           // 0..31
    const int ldc = (tid & 7) * 4;      // 0..28 step 4

    for (int k0 = 0; k0 < K; k0 += TK) {
        // cooperative load + fp32->bf16 convert:  A = X tile (128x32), B = W tile (128x32)
#pragma unroll
        for (int p = 0; p < 4; ++p) {
            int row = ldr + p * 32;
            float4 fa = *(const float4*)(X + (size_t)(mt*TM + row)*K + k0 + ldc);
            bf16x4 ha; ha[0]=(bf16_t)fa.x; ha[1]=(bf16_t)fa.y; ha[2]=(bf16_t)fa.z; ha[3]=(bf16_t)fa.w;
            *(bf16x4*)&lA[row*PITCH + ldc] = ha;
            float4 fb = *(const float4*)(W + (size_t)(nt*TN + row)*K + k0 + ldc);
            bf16x4 hb; hb[0]=(bf16_t)fb.x; hb[1]=(bf16_t)fb.y; hb[2]=(bf16_t)fb.z; hb[3]=(bf16_t)fb.w;
            *(bf16x4*)&lB[row*PITCH + ldc] = hb;
        }
        if (k0 + TK < K) {   // global_prefetch_b8 of next k-slab
            __builtin_prefetch(X + (size_t)(mt*TM + ldr)*K + k0 + TK + ldc, 0, 0);
            __builtin_prefetch(W + (size_t)(nt*TN + ldr)*K + k0 + TK + ldc, 0, 0);
        }
        __syncthreads();

        Frag af;   // A: rows w*16 + (lane%16); k = j + 16g
        {
            int m = w*16 + ln;
            af.h[0] = *(const bf16x8*)&lA[m*PITCH + 16*g];
            af.h[1] = *(const bf16x8*)&lA[m*PITCH + 16*g + 8];
        }
#pragma unroll
        for (int s = 0; s < 8; ++s) {
            Frag bf;   // B: col n = s*16+ln; k = j + 16g  (B[k][n] = W[n][k])
            int n = s*16 + ln;
            bf.h[0] = *(const bf16x8*)&lB[n*PITCH + 16*g];
            bf.h[1] = *(const bf16x8*)&lB[n*PITCH + 16*g + 8];
            acc[s] = wmma_bf16(af.v, bf.v, acc[s]);
        }
        __syncthreads();
    }

#pragma unroll
    for (int s = 0; s < 8; ++s) {
        int n = nt*TN + s*16 + ln;
#pragma unroll
        for (int r = 0; r < 8; ++r) {
            int m = mt*TM + w*16 + 8*g + r;   // C/D layout: vgpr r -> M = r + 8g
            O[(size_t)m * N + n] = (bf16_t)acc[s][r];
        }
    }
}

// ---------------------------------------------------------------------------
// Kernel 2: flash-style attention per (b, h, 128-query block).
// grid: B*H*(S/128) = 512 blocks, 256 threads; wave w owns q rows [w*16, w*16+16).
// K tile (32 keys x 64 d, bf16) is fetched by the Tensor Data Mover directly
// into LDS with hardware padding (4 DWORDs every 32 DWORDs -> pitch 72 bf16),
// issued by wave 0 and published with s_wait_tensorcnt + workgroup barrier,
// overlapping the VALU transpose of the V tile.
// ---------------------------------------------------------------------------
__global__ __launch_bounds__(256) void attn_kernel(
    const bf16_t* __restrict__ Qws, const bf16_t* __restrict__ Kws,
    const bf16_t* __restrict__ Vws, const int* __restrict__ valid_len,
    bf16_t* __restrict__ Cws)
{
    __shared__ alignas(16) bf16_t lK [32 * KPITCH];    // [key][d]   pitch 72 (TDM pad)
    __shared__ alignas(16) bf16_t lVt[64 * PITCH];     // [d][key]   pitch 40 (transposed)
    __shared__ alignas(16) bf16_t lP [8 * 16 * PITCH]; // per-wave P [m][kk] pitch 40

    const int qb = blockIdx.x & 15;
    const int h  = (blockIdx.x >> 4) & (HH - 1);
    const int b  = blockIdx.x >> 8;
    const int vl = valid_len[b];

    const int tid  = threadIdx.x;
    const int lane = tid & 31;
    const int w    = tid >> 5;
    const int g    = lane >> 4;
    const int ln   = lane & 15;
    const int q0   = qb*128 + w*16;

    // Q A-fragments for the two K-halves of hd=64 (kept in registers all along)
    Frag qa[2];
    {
        const bf16_t* qp = Qws + (size_t)(b*SS + q0 + ln)*DD + h*HD;
#pragma unroll
        for (int kh = 0; kh < 2; ++kh) {
            qa[kh].h[0] = *(const bf16x8*)(qp + kh*32 + 16*g);
            qa[kh].h[1] = *(const bf16x8*)(qp + kh*32 + 16*g + 8);
        }
    }

    v8f oacc[4];
#pragma unroll
    for (int s = 0; s < 4; ++s) oacc[s] = vzero8();
    float rmax[8], rsum[8];
#pragma unroll
    for (int r = 0; r < 8; ++r) { rmax[r] = -3.0e38f; rsum[r] = 0.f; }

    const int npair = (vl + 31) >> 5;       // 32-key pairs actually needed
    const int krow8 = tid >> 3;             // coop-load row 0..31
    const int kcol8 = (tid & 7) * 8;        // coop-load col 0..56 step 8
    const unsigned int lK_off = lds_byte_off(&lK[0]);

    for (int kp = 0; kp < npair; ++kp) {
        const int kbase = kp * 32;

        // --- TDM: async DMA of the 32x64 bf16 K tile into LDS (wave 0 issues) ---
        if (w == 0) {
            unsigned long long ga = (unsigned long long)(unsigned long long)(size_t)
                (Kws + (size_t)(b*SS + kbase)*DD + h*HD);
            // D# group 0: count=1 | lds_addr | global_addr | type=2
            u32x4 g0 = { 1u,
                         lK_off,
                         (unsigned int)(ga & 0xFFFFFFFFu),
                         (unsigned int)((ga >> 32) & 0x1FFFFFFu) | (2u << 30) };
            // D# group 1: data_size=2B, pad_enable, pad_interval=32DW, pad_amount=4DW,
            //             tensor_dim0=64, tensor_dim1=32, tile_dim0=64, tile_dim1=32,
            //             tensor_dim0_stride=DD
            i32x8 g1 = { (int)((1u<<16) | (1u<<20) | (4u<<22) | (3u<<25)),
                         (int)(64u << 16),        // tensor_dim0[15:0] in bits 63:48
                         (int)(32u << 16),        // td0 hi=0 | tensor_dim1[15:0]
                         (int)(64u << 16),        // td1 hi=0 | tile_dim0=64
                         (int)32,                 // tile_dim1=32 | tile_dim2=0
                         (int)DD,                 // tensor_dim0_stride[31:0]
                         0,                       // stride0 hi | stride1 lo
                         0 };
            i32x4 gz = { 0, 0, 0, 0 };
#if defined(__clang_major__) && (__clang_major__ >= 23)
            i32x8 gz8 = { 0, 0, 0, 0, 0, 0, 0, 0 };
            __builtin_amdgcn_tensor_load_to_lds(g0, g1, gz, gz, gz8, 0);
#else
            __builtin_amdgcn_tensor_load_to_lds(g0, g1, gz, gz, 0);
#endif
        }

        // cooperative transposed V tile (overlaps the TDM transfer)
        {
            const size_t rowoff = (size_t)(b*SS + kbase + krow8)*DD + h*HD + kcol8;
            bf16x8 vv = *(const bf16x8*)(Vws + rowoff);
#pragma unroll
            for (int e = 0; e < 8; ++e) lVt[(kcol8 + e)*PITCH + krow8] = vv[e];
        }
        if (w == 0) __builtin_amdgcn_s_wait_tensorcnt(0);
        __syncthreads();

        // scores (16q x 32keys) = Q(16x64) @ K^T : two 16-key subtiles, 2 wmmas each
        v8f sc[2];
#pragma unroll
        for (int s = 0; s < 2; ++s) {
            int kr = s*16 + ln;                 // key row in tile (= B col n)
            Frag b0, b1;                        // B[d][key]: slot j -> d = kh*32 + j + 16g
            b0.h[0] = *(const bf16x8*)&lK[kr*KPITCH + 16*g];
            b0.h[1] = *(const bf16x8*)&lK[kr*KPITCH + 16*g + 8];
            b1.h[0] = *(const bf16x8*)&lK[kr*KPITCH + 32 + 16*g];
            b1.h[1] = *(const bf16x8*)&lK[kr*KPITCH + 32 + 16*g + 8];
            v8f z = vzero8();
            z = wmma_bf16(qa[0].v, b0.v, z);
            z = wmma_bf16(qa[1].v, b1.v, z);
            sc[s] = z;
        }

        // scale 1/sqrt(64) + key mask (d2l masked_softmax fill = -1e6)
        const int key0 = kbase + ln, key1 = key0 + 16;
#pragma unroll
        for (int r = 0; r < 8; ++r) {
            float a = sc[0][r] * 0.125f;  if (key0 >= vl) a = -1.0e6f;
            float c = sc[1][r] * 0.125f;  if (key1 >= vl) c = -1.0e6f;
            sc[0][r] = a; sc[1][r] = c;
        }

        // online softmax: rows live as vgpr r (M = r + 8g) across the 16-lane group
        float p0[8], p1[8], corr[8];
#pragma unroll
        for (int r = 0; r < 8; ++r) {
            float mx = fmaxf(sc[0][r], sc[1][r]);
            mx = fmaxf(mx, __shfl_xor(mx, 1));
            mx = fmaxf(mx, __shfl_xor(mx, 2));
            mx = fmaxf(mx, __shfl_xor(mx, 4));
            mx = fmaxf(mx, __shfl_xor(mx, 8));
            float mnew = fmaxf(rmax[r], mx);
            float cr   = __expf(rmax[r] - mnew);
            rmax[r] = mnew;
            float e0 = __expf(sc[0][r] - mnew);
            float e1 = __expf(sc[1][r] - mnew);
            float rs = e0 + e1;
            rs += __shfl_xor(rs, 1);
            rs += __shfl_xor(rs, 2);
            rs += __shfl_xor(rs, 4);
            rs += __shfl_xor(rs, 8);
            rsum[r] = rsum[r] * cr + rs;
            corr[r] = cr; p0[r] = e0; p1[r] = e1;
        }
#pragma unroll
        for (int s = 0; s < 4; ++s)
#pragma unroll
            for (int r = 0; r < 8; ++r) oacc[s][r] *= corr[r];

        // P (D-layout) -> per-wave LDS -> reload in A-layout (same-wave DS is in-order)
        bf16_t* pb = &lP[w * 16 * PITCH];
#pragma unroll
        for (int r = 0; r < 8; ++r) {
            int m = 8*g + r;
            pb[m*PITCH + ln]      = (bf16_t)p0[r];
            pb[m*PITCH + 16 + ln] = (bf16_t)p1[r];
        }
        Frag pa;
        pa.h[0] = *(const bf16x8*)&pb[ln*PITCH + 16*g];
        pa.h[1] = *(const bf16x8*)&pb[ln*PITCH + 16*g + 8];

        // out(16x64) += P(16x32) @ V(32x64): 4 d-subtiles
#pragma unroll
        for (int s = 0; s < 4; ++s) {
            int d = s*16 + ln;
            Frag vb;                            // B[kk][d] = lVt[d][kk], kk = j + 16g
            vb.h[0] = *(const bf16x8*)&lVt[d*PITCH + 16*g];
            vb.h[1] = *(const bf16x8*)&lVt[d*PITCH + 16*g + 8];
            oacc[s] = wmma_bf16(pa.v, vb.v, oacc[s]);
        }
        __syncthreads();
    }

    // finalize: divide by row sums, emit bf16 context for the output projection
#pragma unroll
    for (int r = 0; r < 8; ++r) rsum[r] = 1.0f / rsum[r];
#pragma unroll
    for (int s = 0; s < 4; ++s) {
        int dcol = h*HD + s*16 + ln;
#pragma unroll
        for (int r = 0; r < 8; ++r) {
            int qrow = q0 + 8*g + r;
            Cws[(size_t)(b*SS + qrow)*DD + dcol] = (bf16_t)(oacc[s][r] * rsum[r]);
        }
    }
}

// ---------------------------------------------------------------------------
// Kernel 3: output projection  out = ctx(bf16) @ W_o^T -> fp32 d_out.
// grid: 32*8 = 256 blocks of 256 threads; same tiling as kernel 1.
// ---------------------------------------------------------------------------
__global__ __launch_bounds__(256) void out_proj_kernel(
    const bf16_t* __restrict__ Cws, const float* __restrict__ Wo,
    float* __restrict__ out)
{
    __shared__ alignas(16) bf16_t lA[TM * PITCH];
    __shared__ alignas(16) bf16_t lB[TN * PITCH];

    const int N = DD, K = DD;
    const int mt = blockIdx.x / (N/TN);
    const int nt = blockIdx.x % (N/TN);

    const int tid  = threadIdx.x;
    const int lane = tid & 31;
    const int w    = tid >> 5;
    const int g    = lane >> 4;
    const int ln   = lane & 15;

    v8f acc[8];
#pragma unroll
    for (int s = 0; s < 8; ++s) acc[s] = vzero8();

    const int ar = tid >> 2;            // bf16 A: 64 rows/pass, 2 passes
    const int ac = (tid & 3) * 8;
    const int br = tid >> 3;            // fp32 B: 32 rows/pass, 4 passes
    const int bc = (tid & 7) * 4;

    for (int k0 = 0; k0 < K; k0 += TK) {
#pragma unroll
        for (int p = 0; p < 2; ++p) {
            int row = ar + p * 64;
            bf16x8 v = *(const bf16x8*)(Cws + (size_t)(mt*TM + row)*K + k0 + ac);
            *(bf16x8*)&lA[row*PITCH + ac] = v;
        }
#pragma unroll
        for (int p = 0; p < 4; ++p) {
            int row = br + p * 32;
            float4 f = *(const float4*)(Wo + (size_t)(nt*TN + row)*K + k0 + bc);
            bf16x4 hb; hb[0]=(bf16_t)f.x; hb[1]=(bf16_t)f.y; hb[2]=(bf16_t)f.z; hb[3]=(bf16_t)f.w;
            *(bf16x4*)&lB[row*PITCH + bc] = hb;
        }
        if (k0 + TK < K) {
            __builtin_prefetch(Cws + (size_t)(mt*TM + ar)*K + k0 + TK + ac, 0, 0);
            __builtin_prefetch(Wo  + (size_t)(nt*TN + br)*K + k0 + TK + bc, 0, 0);
        }
        __syncthreads();

        Frag af;
        {
            int m = w*16 + ln;
            af.h[0] = *(const bf16x8*)&lA[m*PITCH + 16*g];
            af.h[1] = *(const bf16x8*)&lA[m*PITCH + 16*g + 8];
        }
#pragma unroll
        for (int s = 0; s < 8; ++s) {
            Frag bf;
            int n = s*16 + ln;
            bf.h[0] = *(const bf16x8*)&lB[n*PITCH + 16*g];
            bf.h[1] = *(const bf16x8*)&lB[n*PITCH + 16*g + 8];
            acc[s] = wmma_bf16(af.v, bf.v, acc[s]);
        }
        __syncthreads();
    }

#pragma unroll
    for (int s = 0; s < 8; ++s) {
        int n = nt*TN + s*16 + ln;
#pragma unroll
        for (int r = 0; r < 8; ++r) {
            int m = mt*TM + w*16 + 8*g + r;
            out[(size_t)m * N + n] = acc[s][r];
        }
    }
}

// ---------------------------------------------------------------------------
extern "C" void kernel_launch(void* const* d_in, const int* in_sizes, int n_in,
                              void* d_out, int out_size, void* d_ws, size_t ws_size,
                              hipStream_t stream) {
    (void)in_sizes; (void)n_in; (void)out_size; (void)ws_size;
    const float* query = (const float*)d_in[0];
    const float* key_t = (const float*)d_in[1];
    const float* value = (const float*)d_in[2];
    const int*   vlen  = (const int*)  d_in[3];
    const float* W_q   = (const float*)d_in[4];
    const float* W_k   = (const float*)d_in[5];
    const float* W_v   = (const float*)d_in[6];
    const float* W_o   = (const float*)d_in[7];

    const size_t mat = (size_t)MM * DD;           // 4096*1024 elements
    bf16_t* Qws = (bf16_t*)d_ws;
    bf16_t* Kws = Qws + mat;
    bf16_t* Vws = Kws + mat;
    bf16_t* Cws = Vws + mat;                      // total 32 MB of bf16 scratch

    qkv_proj_kernel<<<3 * (MM/TM) * (DD/TN), 256, 0, stream>>>(
        query, key_t, value, W_q, W_k, W_v, Qws, Kws, Vws);

    attn_kernel<<<BB * HH * (SS/128), 256, 0, stream>>>(
        Qws, Kws, Vws, vlen, Cws);

    out_proj_kernel<<<(MM/TM) * (DD/TN), 256, 0, stream>>>(
        Cws, W_o, (float*)d_out);
}
